// SetAbstraction_23115513987444
// MI455X (gfx1250) — matrix operand order repaired and checked by reading, not compiled, not run on hardware
//
#include <hip/hip_runtime.h>
#include <hip/hip_bf16.h>

typedef __attribute__((ext_vector_type(16))) _Float16 v16h;
typedef __attribute__((ext_vector_type(8)))  _Float16 v8h;
typedef __attribute__((ext_vector_type(8)))  float    v8f;

#define NPTS 262144
#define DF   64
#define MQ   16384
#define KN   32
#define STR  264   // LDS activation row stride in halves (528B, 16B aligned)

// ---------------- fragment loaders (CDNA5 WMMA 16x16x32 f16 layouts) ----------------

// A-matrix 16x32 f16 per ISA:
// lanes 0-15: row = lane, elems 0..7 -> K k0..k0+7, elems 8..15 -> K k0+16..k0+23
// lanes 16-31: same rows, +8 K offset.
__device__ __forceinline__ v16h lds_afrag(const _Float16* buf, int r0, int k0, int lane) {
    int row  = r0 + (lane & 15);
    int koff = k0 + ((lane & 16) ? 8 : 0);
    const _Float16* p = buf + row * STR + koff;
    v8h lo = *(const v8h*)p;
    v8h hi = *(const v8h*)(p + 16);
    v16h a;
#pragma unroll
    for (int i = 0; i < 8; ++i) { a[i] = lo[i]; a[i + 8] = hi[i]; }
    return a;
}

// B-matrix 32x16 f16 from transposed weights Wt[N][K] in global:
// lane = col (&15); lanes 0-15: elems 0..7 -> K k0..k0+7, elems 8..15 -> K k0+16..k0+23; lanes 16-31: +8.
__device__ __forceinline__ v16h g_bfrag(const _Float16* Wt, int kd, int n0, int k0, int lane) {
    int col  = n0 + (lane & 15);
    int koff = k0 + ((lane & 16) ? 8 : 0);
    const _Float16* p = Wt + (size_t)col * kd + koff;
    v8h lo = *(const v8h*)p;
    v8h hi = *(const v8h*)(p + 16);
    v16h b;
#pragma unroll
    for (int i = 0; i < 8; ++i) { b[i] = lo[i]; b[i + 8] = hi[i]; }
    return b;
}

__device__ __forceinline__ v8f wmma_f(v16h a, v16h b, v8f c) {
    return __builtin_amdgcn_wmma_f32_16x16x32_f16(false, a, false, b, (short)0, c, false, false);
}

// ---------------- Linear + LayerNorm + ReLU layer (one wave = 32 rows) ----------------

template<int KD, int ND, bool HAS_REL>
__device__ __forceinline__ void mlp_layer(
    const _Float16* __restrict__ Wt,                     // [ND][KD] f16 (transposed)
    const float* __restrict__ bias, const float* __restrict__ gamma, const float* __restrict__ beta,
    const float* __restrict__ Wrel, int relStride,       // rel rank-3 part (rows 0..2), if HAS_REL
    const float (*rel)[3],                               // LDS [32][3]
    const _Float16* bufIn, _Float16* bufOut,             // LDS, stride STR
    float* meanArr, float* rstdArr,                      // LDS [32]
    int lane)
{
    constexpr int NKT = KD / 32;
    const int colLane = lane & 15;
    const int rbase   = (lane & 16) ? 8 : 0;

    // hoist all A fragments: they depend only on the k-tile
    v16h a0[NKT], a1[NKT];
#pragma unroll
    for (int kt = 0; kt < NKT; ++kt) {
        a0[kt] = lds_afrag(bufIn, 0,  kt * 32, lane);
        a1[kt] = lds_afrag(bufIn, 16, kt * 32, lane);
    }

    // preload rel (per-row) into registers once
    float relx[16], rely[16], relz[16];
    if (HAS_REL) {
#pragma unroll
        for (int e = 0; e < 8; ++e) {
            int r0 = rbase + e, r1 = 16 + rbase + e;
            relx[e] = rel[r0][0]; rely[e] = rel[r0][1]; relz[e] = rel[r0][2];
            relx[8 + e] = rel[r1][0]; rely[8 + e] = rel[r1][1]; relz[8 + e] = rel[r1][2];
        }
    }

    float rsum[16], rsq[16];
#pragma unroll
    for (int i = 0; i < 16; ++i) { rsum[i] = 0.f; rsq[i] = 0.f; }

#pragma unroll 4
    for (int ct = 0; ct < ND / 16; ++ct) {
        int n0 = ct * 16;
        v8f acc0 = {}; v8f acc1 = {};
#pragma unroll
        for (int kt = 0; kt < NKT; ++kt) {
            v16h b = g_bfrag(Wt, KD, n0, kt * 32, lane);
            acc0 = wmma_f(a0[kt], b, acc0);
            acc1 = wmma_f(a1[kt], b, acc1);
        }
        int col = n0 + colLane;
        float bb = bias[col];
        float wr0 = 0.f, wr1 = 0.f, wr2 = 0.f;
        if (HAS_REL) { wr0 = Wrel[col]; wr1 = Wrel[relStride + col]; wr2 = Wrel[2 * relStride + col]; }
#pragma unroll
        for (int e = 0; e < 8; ++e) {
            float v = acc0[e] + bb;
            if (HAS_REL) v += relx[e] * wr0 + rely[e] * wr1 + relz[e] * wr2;
            bufOut[(rbase + e) * STR + col] = (_Float16)v;
            rsum[e] += v; rsq[e] += v * v;
            float u = acc1[e] + bb;
            if (HAS_REL) u += relx[8 + e] * wr0 + rely[8 + e] * wr1 + relz[8 + e] * wr2;
            bufOut[(16 + rbase + e) * STR + col] = (_Float16)u;
            rsum[8 + e] += u; rsq[8 + e] += u * u;
        }
    }
    // reduce row sums across the 16-lane column group
#pragma unroll
    for (int i = 0; i < 16; ++i) {
#pragma unroll
        for (int msk = 1; msk < 16; msk <<= 1) {
            rsum[i] += __shfl_xor(rsum[i], msk, 32);
            rsq[i]  += __shfl_xor(rsq[i],  msk, 32);
        }
    }
    if (colLane == 0) {
        const float invW = 1.0f / (float)ND;
#pragma unroll
        for (int e = 0; e < 8; ++e) {
            int r0 = rbase + e, r1 = 16 + rbase + e;
            float mu = rsum[e] * invW;
            float va = rsq[e] * invW - mu * mu;
            meanArr[r0] = mu; rstdArr[r0] = rsqrtf(va + 1e-5f);
            mu = rsum[8 + e] * invW;
            va = rsq[8 + e] * invW - mu * mu;
            meanArr[r1] = mu; rstdArr[r1] = rsqrtf(va + 1e-5f);
        }
    }
    __syncthreads();
    // normalize + affine + ReLU: lane handles row == lane, vectorized 8-wide
    {
        float mu = meanArr[lane], rs = rstdArr[lane];
        _Float16* rowp = bufOut + lane * STR;
#pragma unroll 4
        for (int c = 0; c < ND; c += 8) {
            v8h hv = *(const v8h*)(rowp + c);
            float4 ga0 = *(const float4*)(gamma + c);
            float4 ga1 = *(const float4*)(gamma + c + 4);
            float4 bt0 = *(const float4*)(beta + c);
            float4 bt1 = *(const float4*)(beta + c + 4);
            float g[8] = {ga0.x, ga0.y, ga0.z, ga0.w, ga1.x, ga1.y, ga1.z, ga1.w};
            float t[8] = {bt0.x, bt0.y, bt0.z, bt0.w, bt1.x, bt1.y, bt1.z, bt1.w};
            v8h ov;
#pragma unroll
            for (int j = 0; j < 8; ++j) {
                float v = (float)hv[j];
                v = (v - mu) * rs * g[j] + t[j];
                ov[j] = (_Float16)fmaxf(v, 0.f);
            }
            *(v8h*)(rowp + c) = ov;
        }
    }
    __syncthreads();
}

// ---------------- weight pack: f32 [K][N] -> f16 transposed [N][K] ----------------

__global__ void pack_wT(const float* __restrict__ W, _Float16* __restrict__ Wt,
                        int kdim, int ncol, int rowoff, int total) {
    int i = blockIdx.x * blockDim.x + threadIdx.x;
    if (i >= total) return;
    int c = i / kdim;
    int k = i - c * kdim;
    Wt[i] = (_Float16)W[(size_t)(rowoff + k) * ncol + c];
}

// ---------------- main kernel: gather -> MLP x3 -> attention -> weighted sum ----------------

__global__ void __launch_bounds__(128)
sa_main(const float* __restrict__ xyz, const float* __restrict__ features,
        const int* __restrict__ scanidx, const int* __restrict__ group_idx,
        const float* __restrict__ W1, const float* __restrict__ b1,
        const float* __restrict__ g1, const float* __restrict__ be1,
        const float* __restrict__ b2, const float* __restrict__ g2, const float* __restrict__ be2,
        const float* __restrict__ b3, const float* __restrict__ g3, const float* __restrict__ be3,
        const float* __restrict__ Wa1, const float* __restrict__ ba1, const float* __restrict__ Wa2,
        const _Float16* __restrict__ W1t, const _Float16* __restrict__ W2t,
        const _Float16* __restrict__ W3t, const _Float16* __restrict__ Wa1t,
        float* __restrict__ new_xyz, float* __restrict__ weighted)
{
    __shared__ __align__(16) _Float16 s_act[4][2][32][STR];
    __shared__ float s_rel[4][32][3];
    __shared__ float s_mean[4][32];
    __shared__ float s_rstd[4][32];

    int wid  = threadIdx.x >> 5;
    int lane = threadIdx.x & 31;
    int m    = blockIdx.x * 4 + wid;

    _Float16* bufA = &s_act[wid][0][0][0];
    _Float16* bufB = &s_act[wid][1][0][0];
    float (*rel)[3] = s_rel[wid];

    // gather neighbors: lane = neighbor index k
    int si = scanidx[m];
    float nx = xyz[3 * si], ny = xyz[3 * si + 1], nz = xyz[3 * si + 2];
    int gi = group_idx[m * KN + lane];
    rel[lane][0] = xyz[3 * gi]     - nx;
    rel[lane][1] = xyz[3 * gi + 1] - ny;
    rel[lane][2] = xyz[3 * gi + 2] - nz;
    const float4* fp = (const float4*)(features + (size_t)gi * DF);
#pragma unroll
    for (int t = 0; t < DF / 8; ++t) {
        float4 v0 = fp[2 * t];
        float4 v1 = fp[2 * t + 1];
        v8h hv;
        hv[0] = (_Float16)v0.x; hv[1] = (_Float16)v0.y; hv[2] = (_Float16)v0.z; hv[3] = (_Float16)v0.w;
        hv[4] = (_Float16)v1.x; hv[5] = (_Float16)v1.y; hv[6] = (_Float16)v1.z; hv[7] = (_Float16)v1.w;
        *(v8h*)(bufA + lane * STR + t * 8) = hv;
    }
    if (lane == 0) { new_xyz[3 * m] = nx; new_xyz[3 * m + 1] = ny; new_xyz[3 * m + 2] = nz; }
    __syncthreads();

    // per-point MLP stack (rel folded into layer 1 as a rank-3 update)
    mlp_layer< 64, 128, true >(W1t, b1, g1, be1, W1, 128, rel, bufA, bufB, s_mean[wid], s_rstd[wid], lane);
    mlp_layer<128, 128, false>(W2t, b2, g2, be2, nullptr, 0, rel, bufB, bufA, s_mean[wid], s_rstd[wid], lane);
    mlp_layer<128, 256, false>(W3t, b3, g3, be3, nullptr, 0, rel, bufA, bufB, s_mean[wid], s_rstd[wid], lane);
    // h (32x256, f16) now in bufB

    // attention: t = relu([rel,h] @ Wa1 + ba1); score = t @ Wa2 (hidden never stored)
    const int colLane = lane & 15;
    const int rbase   = (lane & 16) ? 8 : 0;

    // hoist the 16 A fragments of h (K = 256)
    v16h ha0[8], ha1[8];
#pragma unroll
    for (int kt = 0; kt < 8; ++kt) {
        ha0[kt] = lds_afrag(bufB, 0,  kt * 32, lane);
        ha1[kt] = lds_afrag(bufB, 16, kt * 32, lane);
    }
    float relx[16], rely[16], relz[16];
#pragma unroll
    for (int e = 0; e < 8; ++e) {
        int r0 = rbase + e, r1 = 16 + rbase + e;
        relx[e] = rel[r0][0]; rely[e] = rel[r0][1]; relz[e] = rel[r0][2];
        relx[8 + e] = rel[r1][0]; rely[8 + e] = rel[r1][1]; relz[8 + e] = rel[r1][2];
    }

    float sacc[16];
#pragma unroll
    for (int i = 0; i < 16; ++i) sacc[i] = 0.f;

#pragma unroll 2
    for (int ct = 0; ct < 16; ++ct) {
        int n0 = ct * 16;
        v8f t0 = {}; v8f t1 = {};
#pragma unroll
        for (int kt = 0; kt < 8; ++kt) {
            v16h b = g_bfrag(Wa1t, 256, n0, kt * 32, lane);
            t0 = wmma_f(ha0[kt], b, t0);
            t1 = wmma_f(ha1[kt], b, t1);
        }
        int col = n0 + colLane;
        float bb = ba1[col], w2s = Wa2[col];
        float wr0 = Wa1[col], wr1 = Wa1[256 + col], wr2 = Wa1[512 + col];
#pragma unroll
        for (int e = 0; e < 8; ++e) {
            float v = t0[e] + bb + relx[e] * wr0 + rely[e] * wr1 + relz[e] * wr2;
            sacc[e] += fmaxf(v, 0.f) * w2s;
            float u = t1[e] + bb + relx[8 + e] * wr0 + rely[8 + e] * wr1 + relz[8 + e] * wr2;
            sacc[8 + e] += fmaxf(u, 0.f) * w2s;
        }
    }
#pragma unroll
    for (int i = 0; i < 16; ++i)
#pragma unroll
        for (int msk = 1; msk < 16; msk <<= 1)
            sacc[i] += __shfl_xor(sacc[i], msk, 32);
    if (colLane == 0) {
#pragma unroll
        for (int e = 0; e < 8; ++e) {
            s_mean[wid][rbase + e]      = sacc[e];
            s_mean[wid][16 + rbase + e] = sacc[8 + e];
        }
    }
    __syncthreads();

    // softmax over the 32 neighbors (ba2 cancels in softmax)
    float sc = s_mean[wid][lane];
    float mx = sc;
#pragma unroll
    for (int msk = 16; msk >= 1; msk >>= 1) mx = fmaxf(mx, __shfl_xor(mx, msk, 32));
    float ex = __expf(sc - mx);
    float sm = ex;
#pragma unroll
    for (int msk = 16; msk >= 1; msk >>= 1) sm += __shfl_xor(sm, msk, 32);
    s_rstd[wid][lane] = ex / sm;
    __syncthreads();

    // weighted sum over neighbors: lane owns 8 output channels
    float acc[8];
#pragma unroll
    for (int j = 0; j < 8; ++j) acc[j] = 0.f;
    int c0 = lane * 8;
#pragma unroll 4
    for (int r = 0; r < 32; ++r) {
        float wr = s_rstd[wid][r];
        v8h hv = *(const v8h*)(bufB + r * STR + c0);
#pragma unroll
        for (int j = 0; j < 8; ++j) acc[j] += wr * (float)hv[j];
    }
    float* dst = weighted + (size_t)m * 256 + c0;
#pragma unroll
    for (int j = 0; j < 8; ++j) dst[j] = acc[j];
}

// ---------------- final out = weighted @ Wo + bo, in place over d_out rows ----------------

__global__ void __launch_bounds__(128)
wo_gemm(float* __restrict__ wbuf, const _Float16* __restrict__ Wot, const float* __restrict__ bo)
{
    int wid  = threadIdx.x >> 5;
    int lane = threadIdx.x & 31;
    int rt   = blockIdx.x * 4 + wid;
    int r0   = rt * 16;

    // preload ALL A fragments (f32 -> f16) before any store: in-place is safe
    v16h a[8];
#pragma unroll
    for (int kt = 0; kt < 8; ++kt) {
        int row  = r0 + (lane & 15);
        int koff = kt * 32 + ((lane & 16) ? 8 : 0);
        const float* p = wbuf + (size_t)row * 256 + koff;
        v16h t;
#pragma unroll
        for (int i = 0; i < 8; ++i) { t[i] = (_Float16)p[i]; t[8 + i] = (_Float16)p[16 + i]; }
        a[kt] = t;
    }
    int colLane = lane & 15, rbase = (lane & 16) ? 8 : 0;
#pragma unroll 4
    for (int ct = 0; ct < 16; ++ct) {
        v8f acc = {};
#pragma unroll
        for (int kt = 0; kt < 8; ++kt) {
            v16h b = g_bfrag(Wot, 256, ct * 16, kt * 32, lane);
            acc = wmma_f(a[kt], b, acc);
        }
        int col = ct * 16 + colLane;
        float bb = bo[col];
#pragma unroll
        for (int e = 0; e < 8; ++e)
            wbuf[(size_t)(r0 + rbase + e) * 256 + col] = acc[e] + bb;
    }
}

// ---------------- host launcher ----------------

extern "C" void kernel_launch(void* const* d_in, const int* in_sizes, int n_in,
                              void* d_out, int out_size, void* d_ws, size_t ws_size,
                              hipStream_t stream) {
    const float* xyz       = (const float*)d_in[0];
    const float* features  = (const float*)d_in[1];
    const int*   scanidx   = (const int*)d_in[2];
    const int*   group_idx = (const int*)d_in[3];
    const float* W1  = (const float*)d_in[4];
    const float* b1  = (const float*)d_in[5];
    const float* g1  = (const float*)d_in[6];
    const float* be1 = (const float*)d_in[7];
    const float* W2  = (const float*)d_in[8];
    const float* b2  = (const float*)d_in[9];
    const float* g2  = (const float*)d_in[10];
    const float* be2 = (const float*)d_in[11];
    const float* W3  = (const float*)d_in[12];
    const float* b3  = (const float*)d_in[13];
    const float* g3  = (const float*)d_in[14];
    const float* be3 = (const float*)d_in[15];
    const float* Wa1 = (const float*)d_in[16];
    const float* ba1 = (const float*)d_in[17];
    const float* Wa2 = (const float*)d_in[18];
    // d_in[19] = ba2 (cancels in softmax)
    const float* Wo  = (const float*)d_in[20];
    const float* bo  = (const float*)d_in[21];

    // packed f16 transposed weights in workspace (376,832 bytes total)
    _Float16* ws   = (_Float16*)d_ws;
    _Float16* W1t  = ws;            //  8192 :  [128][64]   (W1 rows 3..66)
    _Float16* W2t  = ws + 8192;     // 16384 :  [128][128]
    _Float16* W3t  = ws + 24576;    // 32768 :  [256][128]
    _Float16* Wa1t = ws + 57344;    // 65536 :  [256][256]  (Wa1 rows 3..258)
    _Float16* Wot  = ws + 122880;   // 65536 :  [256][256]

    pack_wT<<<(8192  + 255) / 256, 256, 0, stream>>>(W1,  W1t,  64, 128, 3, 8192);
    pack_wT<<<(16384 + 255) / 256, 256, 0, stream>>>(W2,  W2t, 128, 128, 0, 16384);
    pack_wT<<<(32768 + 255) / 256, 256, 0, stream>>>(W3,  W3t, 128, 256, 0, 32768);
    pack_wT<<<(65536 + 255) / 256, 256, 0, stream>>>(Wa1, Wa1t, 256, 256, 3, 65536);
    pack_wT<<<(65536 + 255) / 256, 256, 0, stream>>>(Wo,  Wot, 256, 256, 0, 65536);

    float* outF     = (float*)d_out;
    float* new_xyz  = outF;                      // [M,3]
    float* weighted = outF + (size_t)MQ * 3;     // [M,256] -> becomes final out in place

    sa_main<<<MQ / 4, 128, 0, stream>>>(xyz, features, scanidx, group_idx,
                                        W1, b1, g1, be1, b2, g2, be2, b3, g3, be3,
                                        Wa1, ba1, Wa2,
                                        W1t, W2t, W3t, Wa1t,
                                        new_xyz, weighted);

    wo_gemm<<<MQ / 16 / 4, 128, 0, stream>>>(weighted, Wot, bo);
}
